// MLA_SparseAttention_49100066128379
// MI455X (gfx1250) — matrix operand order repaired and checked by reading, not compiled, not run on hardware
//
#include <hip/hip_runtime.h>

// ---------------- problem constants (B = 1) ----------------
#define S_LEN   1024
#define DIM_    512
#define H_      8
#define DH_     64
#define CBS_    16
#define NCB_    64                    // S/CBS
#define QPROJ_  256
#define KVPROJ_ 341
#define KVPROJP_ 352                  // K padded to multiple of 32
#define NQKV_   (DIM_ + KVPROJ_)      // 853
#define NQKVP_  896                   // padded N (multiple of 64)
#define CKROWS_ 128                   // 65 padded to multiple of 64 (B rows)
#define PROBS_LD 96                   // 65 padded to multiple of 32 (K)
#define CHID_   1024                  // CBS*DH
#define NEGF    (-3.4028235e+38f)
#define EPSF    (1.1920929e-07f)
#define SCALE_  0.125f                // DH^-0.5

typedef __bf16 bf16;
typedef __attribute__((ext_vector_type(16))) __bf16 v16bf;
typedef __attribute__((ext_vector_type(8)))  float  v8f;

union Frag16 { uint4 u[2]; v16bf v; };
union Seg8   { bf16 e[8]; uint4 u; };
union PairU  { bf16 e[2]; unsigned u; };

// =====================================================================
// Batched WMMA GEMM (full-tile only):  C[M,N] = A[M,K] @ B[K,N] (+bias)(relu)
//   Caller guarantees: M % 128 == 0 (grid), N % 64 == 0, K % 32 == 0,
//   lda/ldb multiples of 8. No bounds checks anywhere.
// Block tile 128x64, 8 waves; each wave owns a 32x32 macro-tile =
// 2x2 WMMA tiles -> 4 v_wmma per K-step per wave.
// LDS: A row-major (frag = 2 x ds_load_b128 / lane),
//      B transposed  (frag = 2 x ds_load_b128 / lane, 16B each at k=8*half and k=16+8*half).
// =====================================================================
#define BTM 128
#define BTN 64
#define BTK 32

__global__ __launch_bounds__(256)
void wmma_gemm_bf16(const bf16* __restrict__ A, int lda, long sA,
                    const bf16* __restrict__ B, int ldb, long sB,
                    float* __restrict__ C, int ldc, long sC,
                    bf16* __restrict__ Cb, int ldcb, long sCb,
                    const float* __restrict__ bias,
                    int K, int flags)
{
    const int relu = flags & 1;
    const int trB  = flags & 2;
    const long batch = blockIdx.z;
    A += batch * sA;
    B += batch * sB;

    __shared__ __align__(16) bf16 As[BTM][BTK];   // 8 KB, row-major
    __shared__ __align__(16) bf16 Bt[BTN][BTK];   // 4 KB, transposed [n][k]

    const int tid  = threadIdx.x;
    const int lane = tid & 31;
    const int wave = tid >> 5;
    const int half = lane >> 4;      // 0 or 1
    const int l15  = lane & 15;
    const int wm   = wave >> 1;      // 0..3 -> rows wm*32
    const int wn   = wave & 1;       // 0..1 -> cols wn*32

    const int m0 = blockIdx.y * BTM;
    const int n0 = blockIdx.x * BTN;

    // hoisted staging coordinates
    const int a_r = tid >> 1;              // 0..127
    const int a_k = (tid & 1) * 16;        // 0 or 16
    const int bn_k = (tid >> 3) * 2;       // 0..30 (tid<128), even
    const int bn_n = (tid & 7) * 8;        // 0..56
    const int bt_n = tid >> 2;             // 0..63
    const int bt_k = (tid & 3) * 8;        // 0..24

    const v8f vz = {0.f, 0.f, 0.f, 0.f, 0.f, 0.f, 0.f, 0.f};
    v8f acc[2][2];
    acc[0][0] = vz; acc[0][1] = vz; acc[1][0] = vz; acc[1][1] = vz;

    for (int k0 = 0; k0 < K; k0 += BTK) {
        // ---- stage A: 128x32, 16 contiguous bf16 per thread ----
        const bf16* arow = A + (long)(m0 + a_r) * lda + k0 + a_k;
        uint4 a0 = *(const uint4*)(arow);
        uint4 a1 = *(const uint4*)(arow + 8);
        *(uint4*)(&As[a_r][a_k])     = a0;
        *(uint4*)(&As[a_r][a_k + 8]) = a1;

        // ---- stage B into transposed LDS ----
        if (!trB) {
            if (tid < 128) {
                Seg8 r0, r1;
                r0.u = *(const uint4*)(B + (long)(k0 + bn_k)     * ldb + n0 + bn_n);
                r1.u = *(const uint4*)(B + (long)(k0 + bn_k + 1) * ldb + n0 + bn_n);
                #pragma unroll
                for (int e = 0; e < 8; ++e) {
                    PairU pr; pr.e[0] = r0.e[e]; pr.e[1] = r1.e[e];
                    *(unsigned*)(&Bt[bn_n + e][bn_k]) = pr.u;
                }
            }
        } else {
            uint4 bv = *(const uint4*)(B + (long)(n0 + bt_n) * ldb + k0 + bt_k);
            *(uint4*)(&Bt[bt_n][bt_k]) = bv;
        }
        // prefetch next K tile of A (global_prefetch_b8)
        if (k0 + BTK < K)
            __builtin_prefetch(arow + BTK, 0, 1);
        __syncthreads();

        // ---- fragments: pure ds_load_b128 ----
        Frag16 fa[2], fb[2];
        #pragma unroll
        for (int tm = 0; tm < 2; ++tm) {
            const int ar = wm * 32 + tm * 16 + l15;
            fa[tm].u[0] = *(const uint4*)(&As[ar][half * 8]);
            fa[tm].u[1] = *(const uint4*)(&As[ar][16 + half * 8]);
        }
        #pragma unroll
        for (int tn = 0; tn < 2; ++tn) {
            const int bc = wn * 32 + tn * 16 + l15;
            fb[tn].u[0] = *(const uint4*)(&Bt[bc][half * 8]);
            fb[tn].u[1] = *(const uint4*)(&Bt[bc][16 + half * 8]);
        }
        #pragma unroll
        for (int tm = 0; tm < 2; ++tm)
            #pragma unroll
            for (int tn = 0; tn < 2; ++tn)
                acc[tm][tn] = __builtin_amdgcn_wmma_f32_16x16x32_bf16(
                    false, fa[tm].v, false, fb[tn].v, (short)0, acc[tm][tn], false, false);
        __syncthreads();
    }

    // ---- store: VGPR v -> row = base + 8*half + v ; col lane-striped ----
    #pragma unroll
    for (int tm = 0; tm < 2; ++tm) {
        const int row_base = m0 + wm * 32 + tm * 16 + half * 8;
        #pragma unroll
        for (int tn = 0; tn < 2; ++tn) {
            const int col = n0 + wn * 32 + tn * 16 + l15;
            const float bv = bias ? bias[col] : 0.f;
            #pragma unroll
            for (int v = 0; v < 8; ++v) {
                const int row = row_base + v;
                float val = acc[tm][tn][v] + bv;
                if (relu) val = fmaxf(val, 0.f);
                if (C)  C[batch * sC  + (long)row * ldc  + col] = val;
                if (Cb) Cb[batch * sCb + (long)row * ldcb + col] = (bf16)val;
            }
        }
    }
}

// ---------------- RMSNorm (one block per row) ----------------
__global__ __launch_bounds__(256)
void rmsnorm_kernel(const float* __restrict__ in, const float* __restrict__ w,
                    float* __restrict__ outf, bf16* __restrict__ outb, int dim)
{
    const int row = blockIdx.x;
    const float* x = in + (long)row * dim;
    __shared__ float red[8];
    float ss = 0.f;
    for (int i = threadIdx.x; i < dim; i += blockDim.x) { float v = x[i]; ss += v * v; }
    for (int off = 16; off; off >>= 1) ss += __shfl_xor(ss, off, 32);
    const int lane = threadIdx.x & 31, wv = threadIdx.x >> 5;
    if (lane == 0) red[wv] = ss;
    __syncthreads();
    if (wv == 0) {
        float s = (lane < 8) ? red[lane] : 0.f;
        for (int off = 16; off; off >>= 1) s += __shfl_xor(s, off, 32);
        if (lane == 0) red[0] = s;
    }
    __syncthreads();
    const float inv = rsqrtf(red[0] / (float)dim + EPSF);
    for (int i = threadIdx.x; i < dim; i += blockDim.x) {
        float v = x[i] * inv * w[i];
        if (outf) outf[(long)row * dim + i] = v;
        if (outb) outb[(long)row * dim + i] = (bf16)v;
    }
}

// ---------------- f32 -> bf16 converts ----------------
__global__ void convert_f32_bf16(const float* __restrict__ in, bf16* __restrict__ out, long n)
{
    long i = (long)blockIdx.x * blockDim.x + threadIdx.x;
    if (i < n) out[i] = (bf16)in[i];
}

// zero-padding convert: out is rows_out x ldo; in is rows x cols
__global__ void convert_pad_kernel(const float* __restrict__ in, bf16* __restrict__ out,
                                   int rows, int cols, int ldo, int rows_out)
{
    int idx = blockIdx.x * blockDim.x + threadIdx.x;
    if (idx >= rows_out * ldo) return;
    int r = idx / ldo, c = idx % ldo;
    float v = (r < rows && c < cols) ? in[(long)r * cols + c] : 0.f;
    out[idx] = (bf16)v;
}

// ---------------- build kb/vb blocks (k/v + pos), bf16 ----------------
__global__ void build_blocks_kernel(const float* __restrict__ kvf,
                                    const float* __restrict__ k_pos,
                                    const float* __restrict__ v_pos,
                                    bf16* __restrict__ kb, bf16* __restrict__ vb)
{
    int idx = blockIdx.x * blockDim.x + threadIdx.x;   // H*NCB*CBS*DH = 524288
    if (idx >= H_ * NCB_ * CBS_ * DH_) return;
    int d  = idx & 63;
    int i  = (idx >> 6) & 15;
    int cb = (idx >> 10) & 63;
    int h  = idx >> 16;
    int s  = cb * CBS_ + i;
    float kv = kvf[(long)s * 1024 + h * 64 + d]       + k_pos[(h * 16 + i) * 64 + d];
    float vv = kvf[(long)s * 1024 + 512 + h * 64 + d] + v_pos[(h * 16 + i) * 64 + d];
    long o = (long)(h * NCB_ + cb) * CHID_ + i * 64 + d;
    kb[o] = (bf16)kv;
    vb[o] = (bf16)vv;
}

// ---------------- concat [mem_kv ; ck/cv ; zeros] -> bf16 (per head 128x64) ----------------
__global__ void concat_kernel(const float* __restrict__ ck, const float* __restrict__ cv,
                              const float* __restrict__ mem_kv,
                              bf16* __restrict__ ckf, bf16* __restrict__ cvf)
{
    int idx = blockIdx.x * blockDim.x + threadIdx.x;   // 8*128*64 = 65536
    if (idx >= H_ * CKROWS_ * 64) return;
    int d = idx & 63;
    int j = (idx >> 6) % CKROWS_;
    int h = (idx >> 6) / CKROWS_;
    float kv = 0.f, vv = 0.f;
    if (j == 0) {
        kv = mem_kv[h * 64 + d];
        vv = mem_kv[H_ * 64 + h * 64 + d];
    } else if (j < 65) {
        long o = (long)(h * NCB_ + (j - 1)) * 64 + d;
        kv = ck[o]; vv = cv[o];
    }
    ckf[idx] = (bf16)kv;
    cvf[idx] = (bf16)vv;
}

// ---------------- compressed softmax + importance + top-8 ----------------
__device__ __forceinline__ float cmask_val(const float* row, int j, int s)
{
    int ck_seq = (j == 0) ? -1 : j * CBS_ - 1;
    return (ck_seq < s) ? row[j] * SCALE_ : NEGF;
}

__global__ __launch_bounds__(256)
void csoftmax_topk_kernel(const float* __restrict__ csim, bf16* __restrict__ probs,
                          int* __restrict__ sel)
{
    const int gw   = (blockIdx.x * blockDim.x + threadIdx.x) >> 5;  // h*1024+s
    const int lane = threadIdx.x & 31;
    const int s    = gw & 1023;
    const float* row = csim + (long)gw * CKROWS_;

    // --- softmax over 65 masked keys ---
    float v0 = cmask_val(row, lane, s);
    float v1 = cmask_val(row, lane + 32, s);
    float v2 = (lane == 0) ? cmask_val(row, 64, s) : NEGF;
    float m = fmaxf(fmaxf(v0, v1), v2);
    for (int off = 16; off; off >>= 1) m = fmaxf(m, __shfl_xor(m, off, 32));
    float e0 = __expf(v0 - m), e1 = __expf(v1 - m);
    float e2 = (lane == 0) ? __expf(v2 - m) : 0.f;
    float Z = e0 + e1 + e2;
    for (int off = 16; off; off >>= 1) Z += __shfl_xor(Z, off, 32);
    float inv = 1.f / Z;
    long po = (long)gw * PROBS_LD;
    probs[po + lane]      = (bf16)(e0 * inv);
    probs[po + lane + 32] = (bf16)(e1 * inv);
    probs[po + 64 + lane] = (lane == 0) ? (bf16)(e2 * inv) : (bf16)0.f;  // + zero pad

    // --- importance softmax over [ -1000, csim[1:] ] then drop sentinel ---
    float a = cmask_val(row, 1 + lane, s);     // block index = lane
    float b = cmask_val(row, 33 + lane, s);    // block index = lane+32
    float m2 = fmaxf(fmaxf(a, b), -1000.f);
    for (int off = 16; off; off >>= 1) m2 = fmaxf(m2, __shfl_xor(m2, off, 32));
    float ea = __expf(a - m2), eb = __expf(b - m2);
    float Z2 = ea + eb + ((lane == 0) ? __expf(-1000.f - m2) : 0.f);
    for (int off = 16; off; off >>= 1) Z2 += __shfl_xor(Z2, off, 32);
    float iz = 1.f / Z2;
    float la = ea * iz, lb = eb * iz;

    // --- iterative wave-argmax top-8 (lowest index wins ties) ---
    for (int t = 0; t < 8; ++t) {
        float bv; int bi;
        if (la >= lb) { bv = la; bi = lane; } else { bv = lb; bi = lane + 32; }
        for (int off = 16; off; off >>= 1) {
            float ov = __shfl_xor(bv, off, 32);
            int   oi = __shfl_xor(bi, off, 32);
            if (ov > bv || (ov == bv && oi < bi)) { bv = ov; bi = oi; }
        }
        if (lane == 0) sel[(long)gw * 16 + t] = (bv > 1e-10f) ? bi : -1;
        if (bi == lane)           la = -1.f;
        else if (bi == lane + 32) lb = -1.f;
    }
    if (lane == 0) sel[(long)gw * 16 + 8] = s >> 4;   // own block
}

// ---------------- rotary (in place, q and k halves) ----------------
__global__ void rotary_kernel(float* __restrict__ qf, float* __restrict__ kvf)
{
    int idx = blockIdx.x * blockDim.x + threadIdx.x;   // 1024*8*32 = 262144
    if (idx >= S_LEN * H_ * 32) return;
    int p = idx & 31;
    int h = (idx >> 5) & 7;
    int s = idx >> 8;
    float inv_freq = __expf(-((float)(2 * p) / 64.f) * 9.2103403719761836f); // 10000^-(2p/64)
    float ang = (float)s * inv_freq;
    float sn, cs;
    __sincosf(ang, &sn, &cs);
    long oq = (long)s * 512 + h * 64 + 2 * p;
    float x0 = qf[oq], x1 = qf[oq + 1];
    qf[oq]     = x0 * cs - x1 * sn;
    qf[oq + 1] = x1 * cs + x0 * sn;
    long ok = (long)s * 1024 + h * 64 + 2 * p;
    x0 = kvf[ok]; x1 = kvf[ok + 1];
    kvf[ok]     = x0 * cs - x1 * sn;
    kvf[ok + 1] = x1 * cs + x0 * sn;
}

// ---------------- fine (selected-block) attention: one wave per (h,s) ----------------
__global__ __launch_bounds__(256)
void fine_attn_kernel(const float* __restrict__ qf, const float* __restrict__ kvf,
                      const int* __restrict__ sel, float* __restrict__ f_out)
{
    __shared__ float qsh[8][64];
    __shared__ float psh[8][144];
    const int wv = threadIdx.x >> 5, lane = threadIdx.x & 31;
    const int gw = blockIdx.x * 8 + wv;
    const int h = gw >> 10, s = gw & 1023;

    qsh[wv][lane]      = qf[(long)s * 512 + h * 64 + lane];
    qsh[wv][lane + 32] = qf[(long)s * 512 + h * 64 + lane + 32];
    __syncthreads();

    int sb[9];
    #pragma unroll
    for (int t = 0; t < 9; ++t) sb[t] = sel[(long)gw * 16 + t];

    float sims[5];
    float m = NEGF;
    #pragma unroll
    for (int r = 0; r < 5; ++r) {
        const int slot = lane + 32 * r;
        float sim = NEGF;
        if (slot < 144) {
            const int t = slot >> 4, c = slot & 15;
            const int b = sb[t];
            bool valid = (b >= 0);
            if (t == 8) valid = (c <= (s & 15));     // causal diag in own block
            if (valid) {
                const float* kp = kvf + (long)(b * 16 + c) * 1024 + h * 64;
                float acc = 0.f;
                #pragma unroll 8
                for (int d = 0; d < 64; ++d) acc += qsh[wv][d] * kp[d];
                sim = acc * SCALE_;
            }
        }
        sims[r] = sim;
        m = fmaxf(m, sim);
    }
    for (int off = 16; off; off >>= 1) m = fmaxf(m, __shfl_xor(m, off, 32));
    float Z = 0.f;
    #pragma unroll
    for (int r = 0; r < 5; ++r) {
        const int slot = lane + 32 * r;
        if (slot < 144) {
            float e = __expf(sims[r] - m);
            psh[wv][slot] = e;
            Z += e;
        }
    }
    for (int off = 16; off; off >>= 1) Z += __shfl_xor(Z, off, 32);
    const float inv = 1.f / Z;
    __syncthreads();

    #pragma unroll
    for (int dd = 0; dd < 2; ++dd) {
        const int d = lane + dd * 32;
        float acc = 0.f;
        for (int t = 0; t < 9; ++t) {
            const int b = sb[t];
            if (b < 0) continue;
            const float* vp = kvf + (long)(b * 16) * 1024 + 512 + h * 64 + d;
            #pragma unroll
            for (int c = 0; c < 16; ++c) acc += psh[wv][t * 16 + c] * vp[(long)c * 1024];
        }
        f_out[((long)h * S_LEN + s) * 64 + d] = acc * inv;
    }
}

// ---------------- sliding-window attention (keys s-64..s): one wave per (h,s) ----------------
__global__ __launch_bounds__(256)
void sliding_attn_kernel(const float* __restrict__ qf, const float* __restrict__ kvf,
                         float* __restrict__ s_out)
{
    __shared__ float qsh[8][64];
    __shared__ float psh[8][65];
    const int wv = threadIdx.x >> 5, lane = threadIdx.x & 31;
    const int gw = blockIdx.x * 8 + wv;
    const int h = gw >> 10, s = gw & 1023;

    qsh[wv][lane]      = qf[(long)s * 512 + h * 64 + lane];
    qsh[wv][lane + 32] = qf[(long)s * 512 + h * 64 + lane + 32];
    __syncthreads();

    const int j0 = s - 64;
    float sims[3];
    float m = NEGF;
    #pragma unroll
    for (int r = 0; r < 3; ++r) {
        const int c = lane + 32 * r;
        const int jj = j0 + c;
        float sim = NEGF;
        if (c < 65 && jj >= 0) {
            const float* kp = kvf + (long)jj * 1024 + h * 64;
            float acc = 0.f;
            #pragma unroll 8
            for (int d = 0; d < 64; ++d) acc += qsh[wv][d] * kp[d];
            sim = acc * SCALE_;
        }
        sims[r] = sim;
        m = fmaxf(m, sim);
    }
    for (int off = 16; off; off >>= 1) m = fmaxf(m, __shfl_xor(m, off, 32));
    float Z = 0.f;
    #pragma unroll
    for (int r = 0; r < 3; ++r) {
        const int c = lane + 32 * r;
        if (c < 65) {
            float e = __expf(sims[r] - m);
            psh[wv][c] = e;
            Z += e;
        }
    }
    for (int off = 16; off; off >>= 1) Z += __shfl_xor(Z, off, 32);
    const float inv = 1.f / Z;
    __syncthreads();

    #pragma unroll
    for (int dd = 0; dd < 2; ++dd) {
        const int d = lane + dd * 32;
        float acc = 0.f;
        for (int c = 0; c < 65; ++c) {
            const int jj = j0 + c;
            if (jj >= 0)
                acc += psh[wv][c] * kvf[(long)jj * 1024 + 512 + h * 64 + d];
        }
        s_out[((long)h * S_LEN + s) * 64 + d] = acc * inv;
    }
}

// ---------------- gated combine (adds comb bias here) -> bf16 (s, H*DH) ----------------
__global__ void combine_kernel(const float* __restrict__ g_raw,
                               const float* __restrict__ comb_b,
                               const float* __restrict__ c_out,
                               const float* __restrict__ f_out,
                               const float* __restrict__ s_out,
                               bf16* __restrict__ attn)
{
    int idx = blockIdx.x * blockDim.x + threadIdx.x;   // 1024*512
    if (idx >= S_LEN * DIM_) return;
    int d = idx & 63;
    int h = (idx >> 6) & 7;
    int s = idx >> 9;
    const float* gr = g_raw + (long)s * 64 + h * 3;
    float g0 = 1.f / (1.f + __expf(-(gr[0] + comb_b[h * 3 + 0])));
    float g1 = 1.f / (1.f + __expf(-(gr[1] + comb_b[h * 3 + 1])));
    float g2 = 1.f / (1.f + __expf(-(gr[2] + comb_b[h * 3 + 2])));
    long o = ((long)h * S_LEN + s) * 64 + d;
    float val = g0 * c_out[o] + g1 * f_out[o] + g2 * s_out[o];
    attn[(long)s * 512 + h * 64 + d] = (bf16)val;
}

// =====================================================================
// host side
// =====================================================================
static void launch_gemm(hipStream_t st,
                        const bf16* A, int lda, long sA,
                        const bf16* B, int ldb, long sB,
                        float* C, int ldc, long sC,
                        bf16* Cb, int ldcb, long sCb,
                        const float* bias, int M, int N, int K, int flags, int batch)
{
    dim3 grid(N / BTN, M / BTM, batch);
    wmma_gemm_bf16<<<grid, 256, 0, st>>>(A, lda, sA, B, ldb, sB,
                                         C, ldc, sC, Cb, ldcb, sCb,
                                         bias, K, flags);
}

static void launch_conv(hipStream_t st, const float* in, bf16* out, long n)
{
    convert_f32_bf16<<<(int)((n + 255) / 256), 256, 0, st>>>(in, out, n);
}

static void launch_pad(hipStream_t st, const float* in, bf16* out,
                       int rows, int cols, int ldo, int rows_out)
{
    convert_pad_kernel<<<(rows_out * ldo + 255) / 256, 256, 0, st>>>(
        in, out, rows, cols, ldo, rows_out);
}

extern "C" void kernel_launch(void* const* d_in, const int* in_sizes, int n_in,
                              void* d_out, int out_size, void* d_ws, size_t ws_size,
                              hipStream_t stream)
{
    (void)in_sizes; (void)n_in; (void)out_size;
    const float* inp    = (const float*)d_in[0];
    const float* norm_w = (const float*)d_in[1];
    const float* W_qkv  = (const float*)d_in[2];
    const float* W_dq   = (const float*)d_in[3];
    const float* W_uq   = (const float*)d_in[4];
    const float* q_ln_w = (const float*)d_in[5];
    const float* W_ukv  = (const float*)d_in[6];
    const float* mem_kv = (const float*)d_in[7];
    const float* k_pos  = (const float*)d_in[8];
    const float* v_pos  = (const float*)d_in[9];
    const float* kc_w1  = (const float*)d_in[10];
    const float* kc_b1  = (const float*)d_in[11];
    const float* kc_w2  = (const float*)d_in[12];
    const float* kc_b2  = (const float*)d_in[13];
    const float* vc_w1  = (const float*)d_in[14];
    const float* vc_b1  = (const float*)d_in[15];
    const float* vc_w2  = (const float*)d_in[16];
    const float* vc_b2  = (const float*)d_in[17];
    const float* comb_w = (const float*)d_in[18];
    const float* comb_b = (const float*)d_in[19];
    const float* W_out  = (const float*)d_in[20];
    float* out = (float*)d_out;

    char* p = (char*)d_ws;
    auto carve = [&](size_t bytes) -> char* {
        char* r = p;
        p += (bytes + 255) & ~((size_t)255);
        return r;
    };

    float* x_f32   = (float*)carve((size_t)S_LEN * DIM_ * 4);
    bf16*  x_bf    = (bf16*) carve((size_t)S_LEN * DIM_ * 2);
    bf16*  qkv_bf  = (bf16*) carve((size_t)S_LEN * NQKVP_ * 2);
    float* dq_f32  = (float*)carve((size_t)S_LEN * QPROJ_ * 4);
    bf16*  qn_bf   = (bf16*) carve((size_t)S_LEN * QPROJ_ * 2);
    float* qf_f32  = (float*)carve((size_t)S_LEN * DIM_ * 4);
    bf16*  q_bf    = (bf16*) carve((size_t)S_LEN * DIM_ * 2);
    float* kvf_f32 = (float*)carve((size_t)S_LEN * 1024 * 4);
    bf16*  kb_bf   = (bf16*) carve((size_t)H_ * NCB_ * CHID_ * 2);
    bf16*  vb_bf   = (bf16*) carve((size_t)H_ * NCB_ * CHID_ * 2);
    bf16*  kh_bf   = (bf16*) carve((size_t)H_ * NCB_ * CHID_ * 2);
    bf16*  vh_bf   = (bf16*) carve((size_t)H_ * NCB_ * CHID_ * 2);
    float* ck_f32  = (float*)carve((size_t)H_ * NCB_ * 64 * 4);
    float* cv_f32  = (float*)carve((size_t)H_ * NCB_ * 64 * 4);
    bf16*  ckf_bf  = (bf16*) carve((size_t)H_ * CKROWS_ * 64 * 2);
    bf16*  cvf_bf  = (bf16*) carve((size_t)H_ * CKROWS_ * 64 * 2);
    float* csim    = (float*)carve((size_t)H_ * S_LEN * CKROWS_ * 4);
    bf16*  probs   = (bf16*) carve((size_t)H_ * S_LEN * PROBS_LD * 2);
    int*   sel     = (int*)  carve((size_t)H_ * S_LEN * 16 * 4);
    float* c_out   = (float*)carve((size_t)H_ * S_LEN * 64 * 4);
    float* f_out   = (float*)carve((size_t)H_ * S_LEN * 64 * 4);
    float* s_out   = (float*)carve((size_t)H_ * S_LEN * 64 * 4);
    float* g_raw   = (float*)carve((size_t)S_LEN * 64 * 4);
    bf16*  attn_bf = (bf16*) carve((size_t)S_LEN * DIM_ * 2);
    bf16*  wqkv_b  = (bf16*) carve((size_t)DIM_ * NQKVP_ * 2);
    bf16*  wdq_b   = (bf16*) carve((size_t)DIM_ * QPROJ_ * 2);
    bf16*  wuq_b   = (bf16*) carve((size_t)QPROJ_ * DIM_ * 2);
    bf16*  wukv_b  = (bf16*) carve((size_t)KVPROJP_ * 1024 * 2);
    bf16*  kcw1_b  = (bf16*) carve((size_t)CHID_ * CHID_ * 2);
    bf16*  kcw2_b  = (bf16*) carve((size_t)CHID_ * 64 * 2);
    bf16*  vcw1_b  = (bf16*) carve((size_t)CHID_ * CHID_ * 2);
    bf16*  vcw2_b  = (bf16*) carve((size_t)CHID_ * 64 * 2);
    bf16*  combw_b = (bf16*) carve((size_t)DIM_ * 64 * 2);
    bf16*  wout_b  = (bf16*) carve((size_t)DIM_ * DIM_ * 2);
    if ((size_t)(p - (char*)d_ws) > ws_size) return;

    // --- weight conversions (zero-padded where needed) ---
    launch_pad(stream, W_qkv, wqkv_b, DIM_, NQKV_, NQKVP_, DIM_);        // 512 x 896
    launch_conv(stream, W_dq,  wdq_b,  (long)DIM_ * QPROJ_);
    launch_conv(stream, W_uq,  wuq_b,  (long)QPROJ_ * DIM_);
    launch_pad(stream, W_ukv, wukv_b, KVPROJ_, 1024, 1024, KVPROJP_);    // 352 x 1024
    launch_conv(stream, kc_w1, kcw1_b, (long)CHID_ * CHID_);
    launch_conv(stream, kc_w2, kcw2_b, (long)CHID_ * 64);
    launch_conv(stream, vc_w1, vcw1_b, (long)CHID_ * CHID_);
    launch_conv(stream, vc_w2, vcw2_b, (long)CHID_ * 64);
    launch_pad(stream, comb_w, combw_b, DIM_, 24, 64, DIM_);             // 512 x 64
    launch_conv(stream, W_out, wout_b, (long)DIM_ * DIM_);

    // --- x = rms_norm(inp) ---
    rmsnorm_kernel<<<S_LEN, 256, 0, stream>>>(inp, norm_w, x_f32, x_bf, DIM_);

    // --- qkv = x @ W_qkv (bf16 out, N padded to 896) ---
    launch_gemm(stream, x_bf, DIM_, 0, wqkv_b, NQKVP_, 0,
                nullptr, 0, 0, qkv_bf, NQKVP_, 0, nullptr,
                S_LEN, NQKVP_, DIM_, 0, 1);

    // --- dq = q_lat @ W_dq ---
    launch_gemm(stream, qkv_bf, NQKVP_, 0, wdq_b, QPROJ_, 0,
                dq_f32, QPROJ_, 0, nullptr, 0, 0, nullptr,
                S_LEN, QPROJ_, DIM_, 0, 1);

    // --- qn = rms_norm(dq, q_ln_w) ---
    rmsnorm_kernel<<<S_LEN, 256, 0, stream>>>(dq_f32, q_ln_w, nullptr, qn_bf, QPROJ_);

    // --- qf = qn @ W_uq (f32 + bf16) ---
    launch_gemm(stream, qn_bf, QPROJ_, 0, wuq_b, DIM_, 0,
                qf_f32, DIM_, 0, q_bf, DIM_, 0, nullptr,
                S_LEN, DIM_, QPROJ_, 0, 1);

    // --- kvf = kv_lat @ W_ukv  (K padded to 352; pad cols of qkv are zero) ---
    launch_gemm(stream, qkv_bf + DIM_, NQKVP_, 0, wukv_b, 1024, 0,
                kvf_f32, 1024, 0, nullptr, 0, 0, nullptr,
                S_LEN, 1024, KVPROJP_, 0, 1);

    // --- kb/vb blocks ---
    build_blocks_kernel<<<(H_ * NCB_ * CBS_ * DH_) / 256, 256, 0, stream>>>(
        kvf_f32, k_pos, v_pos, kb_bf, vb_bf);

    // --- compressor MLPs: hidden = relu(kb @ w1 + b1); ck = hidden @ w2 + b2 ---
    launch_gemm(stream, kb_bf, CHID_, 0, kcw1_b, CHID_, 0,
                nullptr, 0, 0, kh_bf, CHID_, 0, kc_b1,
                H_ * NCB_, CHID_, CHID_, 1, 1);
    launch_gemm(stream, kh_bf, CHID_, 0, kcw2_b, 64, 0,
                ck_f32, 64, 0, nullptr, 0, 0, kc_b2,
                H_ * NCB_, 64, CHID_, 0, 1);
    launch_gemm(stream, vb_bf, CHID_, 0, vcw1_b, CHID_, 0,
                nullptr, 0, 0, vh_bf, CHID_, 0, vc_b1,
                H_ * NCB_, CHID_, CHID_, 1, 1);
    launch_gemm(stream, vh_bf, CHID_, 0, vcw2_b, 64, 0,
                cv_f32, 64, 0, nullptr, 0, 0, vc_b2,
                H_ * NCB_, 64, CHID_, 0, 1);

    // --- [mem_kv ; ck/cv ; zeros] -> per-head 128x64 bf16 ---
    concat_kernel<<<(H_ * CKROWS_ * 64) / 256, 256, 0, stream>>>(
        ck_f32, cv_f32, mem_kv, ckf_bf, cvf_bf);

    // --- csim[h] = q[h] @ ck_full[h]^T (batched, transB, N=128) ---
    launch_gemm(stream, q_bf, DIM_, 64, ckf_bf, 64, (long)CKROWS_ * 64,
                csim, CKROWS_, (long)S_LEN * CKROWS_, nullptr, 0, 0, nullptr,
                S_LEN, CKROWS_, 64, 2, H_);

    // --- compressed softmax + importance + top-8 ---
    csoftmax_topk_kernel<<<(H_ * S_LEN) / 8, 256, 0, stream>>>(csim, probs, sel);

    // --- c_out[h] = probs[h] @ cv_full[h] (batched, K=96) ---
    launch_gemm(stream, probs, PROBS_LD, (long)S_LEN * PROBS_LD, cvf_bf, 64, (long)CKROWS_ * 64,
                c_out, 64, (long)S_LEN * 64, nullptr, 0, 0, nullptr,
                S_LEN, 64, PROBS_LD, 0, 1 * H_);

    // --- rotary on q and k (in place) ---
    rotary_kernel<<<(S_LEN * H_ * 32) / 256, 256, 0, stream>>>(qf_f32, kvf_f32);

    // --- fine (selected-block) attention ---
    fine_attn_kernel<<<(H_ * S_LEN) / 8, 256, 0, stream>>>(qf_f32, kvf_f32, sel, f_out);

    // --- sliding-window attention ---
    sliding_attn_kernel<<<(H_ * S_LEN) / 8, 256, 0, stream>>>(qf_f32, kvf_f32, s_out);

    // --- gate logits = x @ comb_w (bias applied in combine) ---
    launch_gemm(stream, x_bf, DIM_, 0, combw_b, 64, 0,
                g_raw, 64, 0, nullptr, 0, 0, nullptr,
                S_LEN, 64, DIM_, 0, 1);

    // --- gated combine -> bf16 ---
    combine_kernel<<<(S_LEN * DIM_) / 256, 256, 0, stream>>>(
        g_raw, comb_b, c_out, f_out, s_out, attn_bf);

    // --- final projection -> d_out ---
    launch_gemm(stream, attn_bf, DIM_, 0, wout_b, DIM_, 0,
                out, DIM_, 0, nullptr, 0, 0, nullptr,
                S_LEN, DIM_, DIM_, 0, 1);
}
